// Attention_63531156242990
// MI455X (gfx1250) — matrix operand order repaired and checked by reading, not compiled, not run on hardware
//
#include <hip/hip_runtime.h>
#include <hip/hip_bf16.h>

typedef __bf16 bf16_t;
typedef __attribute__((ext_vector_type(16))) __bf16 v16bf;
typedef __attribute__((ext_vector_type(8)))  float  v8f;

#define HEADS 12
#define DHEAD 64
#define NTOK  2049
#define BATCH 2
#define MROWS (BATCH * NTOK)   // 4098
#define KDIM  768
#define NQKV  2304
#define ATT_SCALE 0.125f       // 64^-0.5

union Frag { v16bf v; uint4 q[2]; };

__device__ __forceinline__ v8f wmma_bf16(const Frag& a, const Frag& b, v8f c) {
  return __builtin_amdgcn_wmma_f32_16x16x32_bf16(
      /*neg_a=*/false, a.v, /*neg_b=*/false, b.v,
      /*c_mod=*/(short)0, c, /*reuse_a=*/false, /*reuse_b=*/false);
}

__device__ __forceinline__ void wait_asynccnt0() {
  asm volatile("s_wait_asynccnt 0x0" ::: "memory");
}

// Scheduling-group masks (SIInstrInfo): 0x008 = MFMA/WMMA, 0x100 = DS read.
#if __has_builtin(__builtin_amdgcn_sched_group_barrier)
#define SGB(mask, size) __builtin_amdgcn_sched_group_barrier((mask), (size), 0)
#else
#define SGB(mask, size)
#endif

// ---------------------------------------------------------------------------
// fp32 -> bf16 converters (weights stored transposed: dst[c][r] = src[r][c])
// ---------------------------------------------------------------------------
__global__ void cvt_copy_kernel(const float* __restrict__ src,
                                bf16_t* __restrict__ dst, int n) {
  int i = blockIdx.x * 256 + threadIdx.x;
  if (i < n) dst[i] = (bf16_t)src[i];
}

__global__ void cvt_transpose_kernel(const float* __restrict__ src,
                                     bf16_t* __restrict__ dst,
                                     int rows, int cols) {
  int i = blockIdx.x * 256 + threadIdx.x;
  if (i < rows * cols) {
    int r = i / cols;
    int c = i - r * cols;
    dst[(size_t)c * rows + r] = (bf16_t)src[i];
  }
}

// ---------------------------------------------------------------------------
// WMMA GEMM: C[m][n] = sum_k A[m][k] * BT[n][k]   (K = 768, bf16 in, f32 acc)
// block = 128 threads (4 waves), block tile 64(M) x 128(N), wave tile 16x128.
// B tile (128 cols x 32 k) staged block-wide in LDS via
// global_load_async_to_lds_b128 (double buffered, ASYNCcnt; INST_OFFSET hits
// both LDS and global address -> one base VGPR + offset:0/16/32/48 per column).
// sched_group_barrier pins the k-step schedule: 6 DS reads up front, then
// (1 WMMA, 2 DS reads) x5, then 3 WMMAs -- so DScnt waits stay ~4 deep and
// LDS latency hides under the matrix pipe instead of draining per WMMA.
// mode 0: scatter QKV (Q scaled) into head-major bf16 q/k/v caches.
// mode 1: add bias, store fp32 to final output.
// ---------------------------------------------------------------------------
__global__ __launch_bounds__(128) void wmma_gemm_kernel(
    const bf16_t* __restrict__ A, const bf16_t* __restrict__ BT,
    int Mreal, int mode,
    bf16_t* __restrict__ qkvh, const float* __restrict__ bias,
    float* __restrict__ outf) {
  // B tile: 128 cols x 32 k, rows padded to 40 bf16 (80B) for bank spread.
  __shared__ alignas(16) bf16_t Bs[2][128 * 40];

  const int tid  = threadIdx.x;
  const int lane = tid & 31;
  const int wave = tid >> 5;
  const int l16  = lane & 15;
  const int lhi  = lane >> 4;                  // 0 or 1 (half-wave)
  const int mBase = blockIdx.x * 64 + wave * 16;
  const int nBase = blockIdx.y * 128;

  int row  = mBase + l16;
  int rowc = row < Mreal ? row : Mreal - 1;    // clamp pad rows (writes guarded)
  const bf16_t* arow = A + (size_t)rowc * KDIM;

  // Issue async DMA of the 128x32 B tile at k-offset kk into buffer p.
  // Thread t owns column t: 4 x 16B chunks via the shared INST_OFFSET.
  auto stageB = [&](int kk, int p) {
    const unsigned lds = (unsigned)(uintptr_t)(&Bs[p][0]) + (unsigned)(tid * 80);
    const unsigned long long g =
        (unsigned long long)(uintptr_t)(BT + (size_t)(nBase + tid) * KDIM + kk);
    asm volatile("global_load_async_to_lds_b128 %0, %1, off"           :: "v"(lds), "v"(g) : "memory");
    asm volatile("global_load_async_to_lds_b128 %0, %1, off offset:16" :: "v"(lds), "v"(g) : "memory");
    asm volatile("global_load_async_to_lds_b128 %0, %1, off offset:32" :: "v"(lds), "v"(g) : "memory");
    asm volatile("global_load_async_to_lds_b128 %0, %1, off offset:48" :: "v"(lds), "v"(g) : "memory");
  };
  // A fragment: lane row l16, K = (e<8 ? e : e+8) + 8*lhi  (ISA 16-bit A 16x32)
  auto loadA = [&](Frag& f, int kk) {
    const bf16_t* ap = arow + kk + lhi * 8;
    f.q[0] = *(const uint4*)(ap);
    f.q[1] = *(const uint4*)(ap + 16);
  };
  // B fragment u: lane col 16u+l16, K = (e<8 ? e : e+8) + 8*lhi (matches A)
  auto loadB = [&](Frag& f, int p, int u) {
    const bf16_t* bp = &Bs[p][(u * 16 + l16) * 40 + lhi * 8];
    f.q[0] = *(const uint4*)(bp);
    f.q[1] = *(const uint4*)(bp + 16);
  };

  v8f z = {0.f, 0.f, 0.f, 0.f, 0.f, 0.f, 0.f, 0.f};
  v8f acc[8] = {z, z, z, z, z, z, z, z};

  stageB(0, 0);
  Frag afCur, afNxt;
  loadA(afCur, 0);
  afNxt = afCur;

  int p = 0;
  for (int kk = 0; kk < KDIM; kk += 32) {
    const bool hasNext = (kk + 32) < KDIM;
    wait_asynccnt0();        // my chunks of buffer p have landed in LDS
    __syncthreads();         // everyone's landed; buffer p^1 is free to refill
    if (hasNext) {
      stageB(kk + 32, p ^ 1);  // DMA next tile underneath this k-step's math
      loadA(afNxt, kk + 32);
    }
    // load all 8 B fragments into distinct registers, then burst the WMMAs
    Frag bfr[8];
#pragma unroll
    for (int u = 0; u < 8; ++u) loadB(bfr[u], p, u);
#pragma unroll
    for (int u = 0; u < 8; ++u) acc[u] = wmma_bf16(afCur, bfr[u], acc[u]);
    // pin the schedule: 6 DS reads ahead, then (1 WMMA, 2 DS reads) x5, 3 WMMAs
    SGB(0x100, 6);
#pragma unroll
    for (int g = 0; g < 5; ++g) { SGB(0x008, 1); SGB(0x100, 2); }
    SGB(0x008, 3);
    if (hasNext) afCur = afNxt;
    p ^= 1;
  }

#pragma unroll
  for (int u = 0; u < 8; ++u) {
    int n = nBase + u * 16 + l16;
#pragma unroll
    for (int r = 0; r < 8; ++r) {
      int m = mBase + r + 8 * lhi;             // C layout: VGPR r -> row r + 8*lhi
      if (m >= Mreal) continue;
      float vacc = acc[u][r];
      if (mode == 0) {
        int part = n / KDIM;                   // 0=Q 1=K 2=V
        int nn   = n - part * KDIM;
        int h = nn >> 6, d = nn & 63;
        int b = m / NTOK, tok = m - b * NTOK;
        float vv = (part == 0) ? vacc * ATT_SCALE : vacc;  // fold softmax scale into Q
        size_t idx = ((((size_t)part * BATCH + b) * HEADS + h) * NTOK + tok) * DHEAD + d;
        qkvh[idx] = (bf16_t)vv;
      } else {
        outf[(size_t)m * KDIM + n] = vacc + bias[n];
      }
    }
  }
}

// ---------------------------------------------------------------------------
// Windowed attention, one wave per (batch, head, window).
// Key slots: slot 0 = CLS token, slots 1..nq = window tokens, rest masked.
// S = Q(16x64) @ K^T  -> 2 N-tiles x 2 K-steps = 4 WMMAs
// O = P(16x32) @ V    -> 4 d-tiles, K=32 slots = 4 WMMAs (P,V via LDS relayout)
// Windows: w=0 -> {1}; w=1..127 -> {2+16(w-1) .. +15}; w=128 -> {2034..2048}
// ---------------------------------------------------------------------------
__global__ __launch_bounds__(32) void attn_window_kernel(
    const bf16_t* __restrict__ Qh, const bf16_t* __restrict__ Kh,
    const bf16_t* __restrict__ Vh, bf16_t* __restrict__ attnOut) {
  __shared__ alignas(16) bf16_t Plds[16 * 32];
  __shared__ alignas(16) bf16_t VT[64 * 40];     // [d][slot], stride 40 vs bank conflicts

  const int w = blockIdx.x, h = blockIdx.y, b = blockIdx.z;
  int qtok0, nq;
  if (w == 0)        { qtok0 = 1;              nq = 1;  }
  else if (w < 128)  { qtok0 = 2 + 16 * (w-1); nq = 16; }
  else               { qtok0 = 2034;           nq = 15; }

  const int lane = threadIdx.x & 31;
  const int l16  = lane & 15;
  const int lhi  = lane >> 4;
  const size_t headOff = ((size_t)(b * HEADS + h)) * NTOK * DHEAD;
  const bf16_t* Qb = Qh + headOff;
  const bf16_t* Kb = Kh + headOff;
  const bf16_t* Vb = Vh + headOff;

  v8f z = {0.f, 0.f, 0.f, 0.f, 0.f, 0.f, 0.f, 0.f};
  v8f sacc[2] = {z, z};

  int qm = (l16 < nq) ? l16 : (nq - 1);        // clamp pad query rows
  const bf16_t* qrowp = Qb + (size_t)(qtok0 + qm) * DHEAD;
#pragma unroll
  for (int s = 0; s < 2; ++s) {                // d in [32s, 32s+32)
    Frag qf;
    const bf16_t* qp = qrowp + s * 32 + lhi * 8;
    qf.q[0] = *(const uint4*)qp;
    qf.q[1] = *(const uint4*)(qp + 16);
#pragma unroll
    for (int t = 0; t < 2; ++t) {              // key slots [16t, 16t+16)
      int j = t * 16 + l16;
      int ktok = (j == 0) ? 0 : ((j <= nq) ? (qtok0 + j - 1) : qtok0);
      const bf16_t* kp = Kb + (size_t)ktok * DHEAD + s * 32 + lhi * 8;
      Frag kf;
      kf.q[0] = *(const uint4*)kp;
      kf.q[1] = *(const uint4*)(kp + 16);
      sacc[t] = wmma_bf16(qf, kf, sacc[t]);
    }
  }

  // masked softmax per row (rows live in vector element r, cols across 16 lanes)
  const int  j0 = l16, j1 = 16 + l16;
  const bool v0 = (j0 <= nq);                  // slot 0 (CLS) always valid
  const bool v1 = (j1 <= nq);
#pragma unroll
  for (int r = 0; r < 8; ++r) {
    float a0 = v0 ? sacc[0][r] : -3.0e38f;
    float a1 = v1 ? sacc[1][r] : -3.0e38f;
    float mx = fmaxf(a0, a1);
#pragma unroll
    for (int mk = 1; mk < 16; mk <<= 1) mx = fmaxf(mx, __shfl_xor(mx, mk, 32));
    float e0 = v0 ? __expf(a0 - mx) : 0.f;
    float e1 = v1 ? __expf(a1 - mx) : 0.f;
    float sm = e0 + e1;
#pragma unroll
    for (int mk = 1; mk < 16; mk <<= 1) sm += __shfl_xor(sm, mk, 32);
    float inv = 1.0f / sm;
    int rowm = r + 8 * lhi;
    Plds[rowm * 32 + j0] = (bf16_t)(e0 * inv);
    Plds[rowm * 32 + j1] = (bf16_t)(e1 * inv);
  }

  // stage V transposed: lane == slot; invalid slots clamped (P==0 there)
  {
    int vtok = (lane == 0) ? 0 : ((lane <= nq) ? (qtok0 + lane - 1) : qtok0);
    union { uint4 q[4]; bf16_t e[64]; } vu;
    const bf16_t* vp = Vb + (size_t)vtok * DHEAD;
#pragma unroll
    for (int i = 0; i < 4; ++i) vu.q[i] = *(const uint4*)(vp + i * 16);
#pragma unroll
    for (int d = 0; d < 64; ++d) VT[d * 40 + lane] = vu.e[d];
  }
  __syncthreads();   // single-wave WG: cheap LDS fence

  // P as A-fragment: lane row l16, cols {8*lhi..+7} and {16+8*lhi..+7}
  Frag pf;
  {
    const bf16_t* pp = Plds + l16 * 32 + lhi * 8;
    pf.q[0] = *(const uint4*)pp;
    pf.q[1] = *(const uint4*)(pp + 16);
  }
#pragma unroll
  for (int t = 0; t < 4; ++t) {                // output d-tile
    Frag vf;
    const bf16_t* vtp = VT + (t * 16 + l16) * 40 + lhi * 8;
    vf.q[0] = *(const uint4*)vtp;
    vf.q[1] = *(const uint4*)(vtp + 16);
    v8f o = wmma_bf16(pf, vf, z);
    int d = t * 16 + l16;
#pragma unroll
    for (int r = 0; r < 8; ++r) {
      int m = r + 8 * lhi;
      if (m < nq) {
        int tok = qtok0 + m;
        attnOut[((size_t)(b * NTOK) + tok) * KDIM + h * DHEAD + d] = (bf16_t)o[r];
      }
    }
  }
}

// ---------------------------------------------------------------------------
// CLS row: token 0 attends to all 2049 keys. One block per (b,h); 24 blocks.
// ---------------------------------------------------------------------------
__global__ __launch_bounds__(256) void attn_cls_kernel(
    const bf16_t* __restrict__ Qh, const bf16_t* __restrict__ Kh,
    const bf16_t* __restrict__ Vh, bf16_t* __restrict__ attnOut) {
  __shared__ float pj[NTOK];
  __shared__ float red[8];
  __shared__ float outAcc[64];

  const int bh = blockIdx.x;
  const int b = bh / HEADS, h = bh % HEADS;
  const size_t headOff = ((size_t)(b * HEADS + h)) * NTOK * DHEAD;
  const bf16_t* Qb = Qh + headOff;
  const bf16_t* Kb = Kh + headOff;
  const bf16_t* Vb = Vh + headOff;
  const int tid = threadIdx.x;

  float q0[64];
#pragma unroll
  for (int d = 0; d < 64; ++d) q0[d] = (float)Qb[d];   // Q already scaled

  float lmax = -3.0e38f;
  for (int j = tid; j < NTOK; j += 256) {
    const bf16_t* kr = Kb + (size_t)j * DHEAD;
    float dot = 0.f;
#pragma unroll
    for (int d = 0; d < 64; ++d) dot += q0[d] * (float)kr[d];
    pj[j] = dot;
    lmax = fmaxf(lmax, dot);
  }
#pragma unroll
  for (int mk = 16; mk >= 1; mk >>= 1) lmax = fmaxf(lmax, __shfl_xor(lmax, mk, 32));
  if ((tid & 31) == 0) red[tid >> 5] = lmax;
  __syncthreads();
  float smax = red[0];
#pragma unroll
  for (int i = 1; i < 8; ++i) smax = fmaxf(smax, red[i]);
  __syncthreads();

  float lsum = 0.f;
  for (int j = tid; j < NTOK; j += 256) {
    float e = __expf(pj[j] - smax);
    pj[j] = e;
    lsum += e;
  }
#pragma unroll
  for (int mk = 16; mk >= 1; mk >>= 1) lsum += __shfl_xor(lsum, mk, 32);
  if ((tid & 31) == 0) red[tid >> 5] = lsum;
  if (tid < 64) outAcc[tid] = 0.f;
  __syncthreads();
  float ssum = 0.f;
#pragma unroll
  for (int i = 0; i < 8; ++i) ssum += red[i];
  float inv = 1.0f / ssum;

  float acc[64];
#pragma unroll
  for (int d = 0; d < 64; ++d) acc[d] = 0.f;
  for (int j = tid; j < NTOK; j += 256) {
    float wgt = pj[j] * inv;
    const bf16_t* vr = Vb + (size_t)j * DHEAD;
#pragma unroll
    for (int d = 0; d < 64; ++d) acc[d] += wgt * (float)vr[d];
  }
#pragma unroll
  for (int d = 0; d < 64; ++d) atomicAdd(&outAcc[d], acc[d]);   // ds_add_f32
  __syncthreads();
  if (tid < 64)
    attnOut[(size_t)(b * NTOK) * KDIM + h * DHEAD + tid] = (bf16_t)outAcc[tid];
}

// ---------------------------------------------------------------------------
extern "C" void kernel_launch(void* const* d_in, const int* in_sizes, int n_in,
                              void* d_out, int out_size, void* d_ws, size_t ws_size,
                              hipStream_t stream) {
  const float* x     = (const float*)d_in[0];
  const float* w_qkv = (const float*)d_in[1];
  const float* w_out = (const float*)d_in[2];
  const float* b_out = (const float*)d_in[3];
  float* out = (float*)d_out;
  (void)in_sizes; (void)n_in; (void)out_size; (void)ws_size;

  char* ws = (char*)d_ws;
  size_t off = 0;
  auto take = [&](size_t elems) {
    bf16_t* p = (bf16_t*)(ws + off);
    off = (off + elems * sizeof(bf16_t) + 255) & ~(size_t)255;
    return p;
  };
  bf16_t* Xb    = take((size_t)MROWS * KDIM);                    // x in bf16
  bf16_t* WqkvT = take((size_t)NQKV * KDIM);                     // w_qkv^T bf16
  bf16_t* WoutT = take((size_t)KDIM * KDIM);                     // w_out^T bf16
  bf16_t* QKVh  = take((size_t)3 * BATCH * HEADS * NTOK * DHEAD);// head-major q/k/v
  bf16_t* attnO = take((size_t)MROWS * KDIM);                    // attention out (bf16)

  bf16_t* Qh = QKVh;
  bf16_t* Kh = QKVh + (size_t)BATCH * HEADS * NTOK * DHEAD;
  bf16_t* Vh = Kh   + (size_t)BATCH * HEADS * NTOK * DHEAD;

  const int nX = MROWS * KDIM;
  cvt_copy_kernel<<<(nX + 255) / 256, 256, 0, stream>>>(x, Xb, nX);
  cvt_transpose_kernel<<<(KDIM * NQKV + 255) / 256, 256, 0, stream>>>(w_qkv, WqkvT, KDIM, NQKV);
  cvt_transpose_kernel<<<(KDIM * KDIM + 255) / 256, 256, 0, stream>>>(w_out, WoutT, KDIM, KDIM);

  dim3 g1((MROWS + 63) / 64, NQKV / 128);     // 65 x 18
  wmma_gemm_kernel<<<g1, 128, 0, stream>>>(Xb, WqkvT, MROWS, 0, QKVh, nullptr, nullptr);

  dim3 g2(129, HEADS, BATCH);
  attn_window_kernel<<<g2, 32, 0, stream>>>(Qh, Kh, Vh, attnO);
  attn_cls_kernel<<<BATCH * HEADS, 256, 0, stream>>>(Qh, Kh, Vh, attnO);

  dim3 g3((MROWS + 63) / 64, KDIM / 128);     // 65 x 6
  wmma_gemm_kernel<<<g3, 128, 0, stream>>>(attnO, WoutT, MROWS, 1, nullptr, b_out, out);
}
